// SASRec_7876970021383
// MI455X (gfx1250) — matrix operand order repaired and checked by reading, not compiled, not run on hardware
//
#include <hip/hip_runtime.h>
#include <hip/hip_bf16.h>

// ---------------- problem constants ----------------
#define B_   32
#define S_   512
#define H_   512
#define L_   2
#define NH_  8
#define DH_  64
#define M_   (B_ * S_)          // 16384 rows
#define KSTEP 64
#define SQRT_H 22.627416997969522f

typedef __bf16 bf16_t;
typedef __attribute__((ext_vector_type(16))) __bf16 v16bf;
typedef __attribute__((ext_vector_type(8)))  __bf16 v8bf;
typedef __attribute__((ext_vector_type(8)))  float  v8f;
typedef __attribute__((ext_vector_type(4)))  float  v4f;

__device__ __forceinline__ bf16_t f2bf(float f) {
    union { float f; unsigned u; } x; x.f = f;
    unsigned r = (x.u + 0x7FFFu + ((x.u >> 16) & 1u)) >> 16;
    unsigned short us = (unsigned short)r;
    bf16_t b; __builtin_memcpy(&b, &us, 2); return b;
}

union FragBF { v16bf v; v8bf h[2]; };

// A-fragment (16x32 bf16, row-major source, lda in elements).
// ISA 7.12.2: lane holds row M=lane&15; K chunks at kb=(lane>=16?8:0), kb+16.
__device__ __forceinline__ v16bf load_a_frag(const bf16_t* A, int lda, int lane) {
    int m  = lane & 15;
    int kb = (lane >> 4) * 8;
    const bf16_t* p = A + (size_t)m * lda + kb;
    FragBF f;
    f.h[0] = *(const v8bf*)(p);
    f.h[1] = *(const v8bf*)(p + 16);
    return f.v;
}

// B-fragment (32x16 bf16) from N-major / K-contiguous source Bt[n][k].
// Lane holds col N=lane&15; K chunks at kb=(lane>=16?8:0), kb+16.
__device__ __forceinline__ v16bf load_b_frag(const bf16_t* Bt, int ldb, int lane) {
    int n  = lane & 15;
    int kb = (lane >> 4) * 8;
    const bf16_t* p = Bt + (size_t)n * ldb + kb;
    FragBF f;
    f.h[0] = *(const v8bf*)(p);
    f.h[1] = *(const v8bf*)(p + 16);
    return f.v;
}

#define WMMA_BF16(a, b, c) \
    __builtin_amdgcn_wmma_f32_16x16x32_bf16(false, (a), false, (b), (short)0, (c), false, false)

// ---------------- weight prep (ONE launch): fp32 W[K][N] -> bf16 Wt[N][K], 12 mats ----
__global__ __launch_bounds__(256)
void prep_weights_kernel(const float* __restrict__ wq, const float* __restrict__ wk,
                         const float* __restrict__ wv, const float* __restrict__ wo,
                         const float* __restrict__ w1, const float* __restrict__ w2,
                         bf16_t* __restrict__ wts) {
    int idx = blockIdx.x * 256 + threadIdx.x;     // 12 * H * H total
    int mat = idx >> 18;                          // H*H = 2^18
    int rem = idx & ((1 << 18) - 1);
    int n = rem >> 9, k = rem & 511;
    int l = mat / 6, j = mat - 6 * l;
    const float* srcs[6] = { wq, wk, wv, wo, w1, w2 };
    const float* W = srcs[j] + (size_t)l * H_ * H_;
    wts[idx] = f2bf(W[(size_t)k * H_ + n]);
}

// ---------------- embedding: x = item_emb[seq]*sqrt(H) + pos_emb ----------------
__global__ __launch_bounds__(128)
void embed_kernel(const int* __restrict__ seq, const int* __restrict__ seq_len,
                  const float* __restrict__ item_emb, const float* __restrict__ pos_emb,
                  float* __restrict__ x) {
    int bs = blockIdx.x;                 // 0 .. B*S-1
    int b = bs / S_, s = bs - b * S_;
    int item = seq[bs];
    int pos = (s < seq_len[b]) ? s : 0;
    const v4f* ie = (const v4f*)(item_emb + (size_t)item * H_);
    const v4f* pe = (const v4f*)(pos_emb  + (size_t)pos  * H_);
    v4f* xo = (v4f*)(x + (size_t)bs * H_);
    int t = threadIdx.x;                 // H/4 = 128
    v4f a = ie[t], p = pe[t], r;
#pragma unroll
    for (int e = 0; e < 4; ++e) r[e] = a[e] * SQRT_H + p[e];
    xo[t] = r;
}

// ---------------- LayerNorm (fp32 in) -> bf16 out, one wave per row ----------------
__global__ __launch_bounds__(256)
void ln_bf16_kernel(const float* __restrict__ x, const float* __restrict__ w,
                    const float* __restrict__ b, bf16_t* __restrict__ out, int rows) {
    int lane = threadIdx.x & 31;
    int row  = blockIdx.x * 8 + (threadIdx.x >> 5);
    if (row >= rows) return;
    const float* xr = x + (size_t)row * H_;
    float vals[16], s = 0.f, ss = 0.f;
    const v4f* xv = (const v4f*)(xr + lane * 16);
#pragma unroll
    for (int t = 0; t < 4; ++t) {
        v4f q = xv[t];
#pragma unroll
        for (int e = 0; e < 4; ++e) { float f = q[e]; vals[t*4+e] = f; s += f; ss += f*f; }
    }
#pragma unroll
    for (int m = 1; m < 32; m <<= 1) { s += __shfl_xor(s, m, 32); ss += __shfl_xor(ss, m, 32); }
    float mean = s * (1.0f / H_);
    float var  = ss * (1.0f / H_) - mean * mean;
    float rstd = rsqrtf(var + 1e-8f);
    bf16_t* orow = out + (size_t)row * H_;
#pragma unroll
    for (int t = 0; t < 16; ++t) {
        int hc = lane * 16 + t;
        orow[hc] = f2bf((vals[t] - mean) * rstd * w[hc] + b[hc]);
    }
}

// ---------------- final LN + gather last valid position -> fp32 [B,H] ----------------
__global__ __launch_bounds__(256)
void final_ln_gather_kernel(const float* __restrict__ x, const int* __restrict__ seq_len,
                            const float* __restrict__ w, const float* __restrict__ b,
                            float* __restrict__ out) {
    int lane = threadIdx.x & 31;
    int row  = blockIdx.x * 8 + (threadIdx.x >> 5);    // batch index
    if (row >= B_) return;
    int s = seq_len[row] - 1;
    const float* xr = x + ((size_t)(row * S_ + s)) * H_;
    float vals[16], sm = 0.f, ss = 0.f;
    const v4f* xv = (const v4f*)(xr + lane * 16);
#pragma unroll
    for (int t = 0; t < 4; ++t) {
        v4f q = xv[t];
#pragma unroll
        for (int e = 0; e < 4; ++e) { float f = q[e]; vals[t*4+e] = f; sm += f; ss += f*f; }
    }
#pragma unroll
    for (int m = 1; m < 32; m <<= 1) { sm += __shfl_xor(sm, m, 32); ss += __shfl_xor(ss, m, 32); }
    float mean = sm * (1.0f / H_);
    float var  = ss * (1.0f / H_) - mean * mean;
    float rstd = rsqrtf(var + 1e-8f);
#pragma unroll
    for (int t = 0; t < 16; ++t) {
        int hc = lane * 16 + t;
        out[(size_t)row * H_ + hc] = (vals[t] - mean) * rstd * w[hc] + b[hc];
    }
}

// ---------------- WMMA GEMM with async-to-LDS double buffering ----------------
// C[M,N] = A[M,K](bf16) x Wt[N,K](bf16) + bias
// MODE 0: bf16 out (+bias)           MODE 1: bf16 out (+bias, ReLU)
// MODE 2: fp32 out = resid+acc+bias  MODE 3: bf16 out (+bias) transposed to [B,NH,DH,S]
template <int MODE>
__global__ __launch_bounds__(256)
void gemm_wmma(const bf16_t* __restrict__ A, const bf16_t* __restrict__ Bt,
               const float* __restrict__ bias, const float* __restrict__ resid,
               bf16_t* __restrict__ outB, float* __restrict__ outF,
               int M, int N, int K) {
    __shared__ bf16_t As[2][128 * KSTEP];   // 2 x 16 KB
    __shared__ bf16_t Bs[2][64 * KSTEP];    // 2 x  8 KB
    const int tid  = threadIdx.x;
    const int lane = tid & 31;
    const int wave = tid >> 5;
    const int wr = wave & 3;               // 4 wave-rows * 32 = 128 M
    const int wc = wave >> 2;              // 2 wave-cols * 32 = 64  N
    const int blocks_n = N >> 6;
    const int bm = (blockIdx.x / blocks_n) << 7;
    const int bn = (blockIdx.x % blocks_n) << 6;

    const bf16_t* Ab = A  + (size_t)bm * K;
    const bf16_t* Bb = Bt + (size_t)bn * K;

    // async stage of one 64-wide K-slab into LDS buffer `buf` (6 b128 per thread)
    auto stage = [&](int buf, int k0) {
#pragma unroll
        for (int r = 0; r < 4; ++r) {           // A: 128 rows x 64 cols = 1024 chunks
            int c   = tid + 256 * r;
            int row = c >> 3, cc = c & 7;
            unsigned lds = (unsigned)(uintptr_t)&As[buf][row * KSTEP + cc * 8];
            const bf16_t* g = Ab + (size_t)row * K + k0 + cc * 8;
            asm volatile("global_load_async_to_lds_b128 %0, %1, off"
                         :: "v"(lds), "v"(g) : "memory");
        }
#pragma unroll
        for (int r = 0; r < 2; ++r) {           // B: 64 rows x 64 cols = 512 chunks
            int c   = tid + 256 * r;
            int row = c >> 3, cc = c & 7;
            unsigned lds = (unsigned)(uintptr_t)&Bs[buf][row * KSTEP + cc * 8];
            const bf16_t* g = Bb + (size_t)row * K + k0 + cc * 8;
            asm volatile("global_load_async_to_lds_b128 %0, %1, off"
                         :: "v"(lds), "v"(g) : "memory");
        }
    };

    v8f zero = {};
    v8f acc00 = zero, acc01 = zero, acc10 = zero, acc11 = zero;

    stage(0, 0);
    const int nsteps = K / KSTEP;
    for (int s = 0; s < nsteps; ++s) {
        int buf = s & 1;
        if (s + 1 < nsteps) {
            stage(buf ^ 1, (s + 1) * KSTEP);
            asm volatile("s_wait_asynccnt 0x6" ::: "memory");   // current slab retired
        } else {
            asm volatile("s_wait_asynccnt 0x0" ::: "memory");
        }
        __syncthreads();
        const bf16_t* Aw = &As[buf][(wr * 32) * KSTEP];
        const bf16_t* Bw = &Bs[buf][(wc * 32) * KSTEP];
#pragma unroll
        for (int kk = 0; kk < KSTEP; kk += 32) {
            v16bf a0 = load_a_frag(Aw + kk,              KSTEP, lane);
            v16bf a1 = load_a_frag(Aw + 16 * KSTEP + kk, KSTEP, lane);
            v16bf b0 = load_b_frag(Bw + kk,              KSTEP, lane);
            v16bf b1 = load_b_frag(Bw + 16 * KSTEP + kk, KSTEP, lane);
            acc00 = WMMA_BF16(a0, b0, acc00);
            acc01 = WMMA_BF16(a0, b1, acc01);
            acc10 = WMMA_BF16(a1, b0, acc10);
            acc11 = WMMA_BF16(a1, b1, acc11);
        }
        __syncthreads();   // all waves done reading before next-stage overwrite
    }

    // C layout: lane holds col n=lane&15, rows (lane/16)*8 + v
    const int m0 = bm + wr * 32;
    const int n0 = bn + wc * 32;
    const int cn = lane & 15;
    const int rg = (lane >> 4) * 8;
    v8f accs[2][2] = { { acc00, acc01 }, { acc10, acc11 } };
#pragma unroll
    for (int j = 0; j < 2; ++j) {
        int n = n0 + j * 16 + cn;
        float bvv = bias ? bias[n] : 0.0f;
#pragma unroll
        for (int i = 0; i < 2; ++i) {
            int mb = m0 + i * 16 + rg;
            if (MODE == 3) {
                // transposed store: n -> (h,d), rows mb..mb+7 -> consecutive s
                int hh = n >> 6, d = n & 63;
                int bb = mb >> 9, s2 = mb & 511;
                v8bf ov;
#pragma unroll
                for (int v = 0; v < 8; ++v) ov[v] = f2bf(accs[i][j][v] + bvv);
                *(v8bf*)&outB[((size_t)((bb * NH_ + hh) * DH_ + d)) * S_ + s2] = ov;
            } else {
#pragma unroll
                for (int v = 0; v < 8; ++v) {
                    int m = mb + v;
                    float val = accs[i][j][v] + bvv;
                    if (MODE == 1) val = fmaxf(val, 0.0f);
                    if (MODE == 2) {
                        outF[(size_t)m * N + n] = resid[(size_t)m * N + n] + val;
                    } else {
                        outB[(size_t)m * N + n] = f2bf(val);
                    }
                }
            }
        }
    }
}

// ---------------- flash attention, fully transposed (S^T = K Q^T, O^T = V^T P^T) ----
// One wave per 16-query tile, 64 keys per iteration. Lane column == query, so
// softmax stats are per-lane scalars: one shfl_xor(16) each for max and sum.
__global__ __launch_bounds__(256)
void attention_kernel(const bf16_t* __restrict__ q, const bf16_t* __restrict__ k,
                      const bf16_t* __restrict__ vt, const int* __restrict__ seq_len,
                      bf16_t* __restrict__ out) {
    __shared__ bf16_t pbuf[8][16 * 64];   // per-wave P staging [query][key]
    const int lane = threadIdx.x & 31;
    const int wave = threadIdx.x >> 5;
    int bid = blockIdx.x;                 // B * NH * 4 blocks
    int qc = bid & 3;
    int h  = (bid >> 2) & 7;
    int b  = bid >> 5;
    int qbase = (qc * 8 + wave) * 16;
    int slen = seq_len[b];

    const bf16_t* qp = q  + ((size_t)(b * S_ + qbase)) * H_ + h * DH_;
    const bf16_t* kp = k  + ((size_t)b * S_) * H_ + h * DH_;
    const bf16_t* vp = vt + ((size_t)(b * NH_ + h)) * DH_ * S_;
    bf16_t*       op = out + ((size_t)(b * S_ + qbase)) * H_ + h * DH_;

    const int cn = lane & 15;             // this lane's query column
    const int rg = (lane >> 4) * 8;
    const int qi = qbase + cn;
    const int kcap0 = ((qi < slen - 1) ? qi : slen - 1) - rg;  // last valid key - rg

    v8f zero = {};
    v8f oacc[4] = { zero, zero, zero, zero };   // O^T: d = nt*16+rg+v, query cn
    float mold = -3.0e38f, lsum = 0.0f;

    // Q as loop-invariant B-fragments: B[d][query]
    v16bf bq0 = load_b_frag(qp,      H_, lane);
    v16bf bq1 = load_b_frag(qp + 32, H_, lane);

    int nkb = (qbase + 16 + 63) >> 6;     // causal: only key blocks <= query tile
    for (int kb = 0; kb < nkb; ++kb) {
        int kt = kb * 64;
        // ---- all 8 K A-fragments first, then 8 WMMAs (staged loadcnt waits) ----
        v16bf ak[8];
#pragma unroll
        for (int t = 0; t < 4; ++t) {
            ak[2*t]   = load_a_frag(kp + (size_t)(kt + t*16) * H_,      H_, lane);
            ak[2*t+1] = load_a_frag(kp + (size_t)(kt + t*16) * H_ + 32, H_, lane);
        }
        v8f st[4] = { zero, zero, zero, zero };
#pragma unroll
        for (int t = 0; t < 4; ++t) {
            st[t] = WMMA_BF16(ak[2*t],   bq0, st[t]);
            st[t] = WMMA_BF16(ak[2*t+1], bq1, st[t]);
        }
        // ---- scale + mask via single bound: valid iff t*16+v <= kcap ----
        int kcap = kcap0 - kt;
        float sv[4][8];
        float mx = -3.0e38f;
#pragma unroll
        for (int t = 0; t < 4; ++t)
#pragma unroll
            for (int v = 0; v < 8; ++v) {
                float xx = st[t][v] * 0.125f;
                sv[t][v] = (t * 16 + v <= kcap) ? xx : -3.0e38f;
                mx = fmaxf(mx, sv[t][v]);
            }
        mx = fmaxf(mx, __shfl_xor(mx, 16, 32));
        float mnew = fmaxf(mold, mx);
        float scl  = __expf(mold - mnew);
        mold = mnew;
        float rs = 0.f;
#pragma unroll
        for (int t = 0; t < 4; ++t) {
            v8bf pv;
#pragma unroll
            for (int v = 0; v < 8; ++v) {
                float e = __expf(sv[t][v] - mnew);
                rs += e;
                pv[v] = f2bf(e);
            }
            *(v8bf*)&pbuf[wave][cn * 64 + t * 16 + rg] = pv;   // ds_store_b128
        }
        rs += __shfl_xor(rs, 16, 32);
        lsum = lsum * scl + rs;
#pragma unroll
        for (int nt = 0; nt < 4; ++nt)
#pragma unroll
            for (int v = 0; v < 8; ++v) oacc[nt][v] *= scl;   // scl per-lane scalar

        asm volatile("s_wait_dscnt 0" ::: "memory");
        // ---- O^T += V^T(16d x 64k) x P^T over two 32-key chunks ----
        v16bf bp0 = load_b_frag(&pbuf[wave][0],  64, lane);
        v16bf bp1 = load_b_frag(&pbuf[wave][32], 64, lane);
        v16bf av[8];
#pragma unroll
        for (int nt = 0; nt < 4; ++nt) {
            av[2*nt]   = load_a_frag(vp + (size_t)(nt * 16) * S_ + kt,      S_, lane);
            av[2*nt+1] = load_a_frag(vp + (size_t)(nt * 16) * S_ + kt + 32, S_, lane);
        }
#pragma unroll
        for (int nt = 0; nt < 4; ++nt) {
            oacc[nt] = WMMA_BF16(av[2*nt],   bp0, oacc[nt]);
            oacc[nt] = WMMA_BF16(av[2*nt+1], bp1, oacc[nt]);
        }
        asm volatile("s_wait_dscnt 0" ::: "memory");          // WAR before next stores
    }
    // ---- epilogue: out[qbase+cn][d] = O^T / l, vectorized 16B stores ----
    float rl = 1.0f / lsum;
#pragma unroll
    for (int nt = 0; nt < 4; ++nt) {
        v8bf ov;
#pragma unroll
        for (int v = 0; v < 8; ++v) ov[v] = f2bf(oacc[nt][v] * rl);
        *(v8bf*)(op + (size_t)cn * H_ + nt * 16 + rg) = ov;
    }
}

// ---------------- host driver ----------------
extern "C" void kernel_launch(void* const* d_in, const int* in_sizes, int n_in,
                              void* d_out, int out_size, void* d_ws, size_t ws_size,
                              hipStream_t stream) {
    const int*   seq      = (const int*)d_in[0];
    const int*   seq_len  = (const int*)d_in[1];
    const float* item_emb = (const float*)d_in[2];
    const float* pos_emb  = (const float*)d_in[3];
    const float* wq = (const float*)d_in[4];   const float* bq = (const float*)d_in[5];
    const float* wk = (const float*)d_in[6];   const float* bk = (const float*)d_in[7];
    const float* wv = (const float*)d_in[8];   const float* bv = (const float*)d_in[9];
    const float* wo = (const float*)d_in[10];  const float* bo = (const float*)d_in[11];
    const float* w1 = (const float*)d_in[12];  const float* b1 = (const float*)d_in[13];
    const float* w2 = (const float*)d_in[14];  const float* b2 = (const float*)d_in[15];
    const float* ln1w = (const float*)d_in[16]; const float* ln1b = (const float*)d_in[17];
    const float* ln2w = (const float*)d_in[18]; const float* ln2b = (const float*)d_in[19];
    const float* lnfw = (const float*)d_in[20]; const float* lnfb = (const float*)d_in[21];
    (void)in_sizes; (void)n_in; (void)out_size; (void)ws_size;

    char* ws = (char*)d_ws;
    size_t off = 0;
    auto take = [&](size_t bytes) -> char* {
        char* p = ws + off;
        off = (off + bytes + 255) & ~(size_t)255;
        return p;
    };
    float*  x   = (float*) take((size_t)M_ * H_ * 4);    // fp32 residual stream
    bf16_t* hb  = (bf16_t*)take((size_t)M_ * H_ * 2);    // LN output (GEMM input)
    bf16_t* qb  = (bf16_t*)take((size_t)M_ * H_ * 2);    // Q  (reused as FFN mid)
    bf16_t* kb_ = (bf16_t*)take((size_t)M_ * H_ * 2);    // K
    bf16_t* aob = (bf16_t*)take((size_t)M_ * H_ * 2);    // attention output
    bf16_t* vtb = (bf16_t*)take((size_t)M_ * H_ * 2);    // V, written transposed [B,NH,DH,S]
    bf16_t* wts = (bf16_t*)take((size_t)12 * H_ * H_ * 2);

    // weight prep: all 12 matrices in one launch
    prep_weights_kernel<<<12 * (H_*H_) / 256, 256, 0, stream>>>(wq, wk, wv, wo, w1, w2, wts);

    embed_kernel<<<B_ * S_, 128, 0, stream>>>(seq, seq_len, item_emb, pos_emb, x);

    const int ggrid = (M_ / 128) * (H_ / 64);   // 1024 blocks
    for (int l = 0; l < L_; ++l) {
        bf16_t* Wq = wts + (size_t)(l*6 + 0) * H_ * H_;
        bf16_t* Wk = wts + (size_t)(l*6 + 1) * H_ * H_;
        bf16_t* Wv = wts + (size_t)(l*6 + 2) * H_ * H_;
        bf16_t* Wo = wts + (size_t)(l*6 + 3) * H_ * H_;
        bf16_t* W1 = wts + (size_t)(l*6 + 4) * H_ * H_;
        bf16_t* W2 = wts + (size_t)(l*6 + 5) * H_ * H_;

        ln_bf16_kernel<<<M_/8, 256, 0, stream>>>(x, ln1w + l*H_, ln1b + l*H_, hb, M_);
        gemm_wmma<0><<<ggrid, 256, 0, stream>>>(hb, Wq, bq + l*H_, nullptr, qb,  nullptr, M_, H_, H_);
        gemm_wmma<0><<<ggrid, 256, 0, stream>>>(hb, Wk, bk + l*H_, nullptr, kb_, nullptr, M_, H_, H_);
        gemm_wmma<3><<<ggrid, 256, 0, stream>>>(hb, Wv, bv + l*H_, nullptr, vtb, nullptr, M_, H_, H_);
        attention_kernel<<<B_*NH_*4, 256, 0, stream>>>(qb, kb_, vtb, seq_len, aob);
        gemm_wmma<2><<<ggrid, 256, 0, stream>>>(aob, Wo, bo + l*H_, x, nullptr, x, M_, H_, H_);
        ln_bf16_kernel<<<M_/8, 256, 0, stream>>>(x, ln2w + l*H_, ln2b + l*H_, hb, M_);
        gemm_wmma<1><<<ggrid, 256, 0, stream>>>(hb, W1, b1 + l*H_, nullptr, qb, nullptr, M_, H_, H_);
        gemm_wmma<2><<<ggrid, 256, 0, stream>>>(qb, W2, b2 + l*H_, x, nullptr, x, M_, H_, H_);
    }

    final_ln_gather_kernel<<<(B_ + 7) / 8, 256, 0, stream>>>(x, seq_len, lnfw, lnfb, (float*)d_out);
}
